// SoftAttentionAlignment_15779709846003
// MI455X (gfx1250) — matrix operand order repaired and checked by reading, not compiled, not run on hardware
//
#include <hip/hip_runtime.h>
#include <math.h>

typedef __attribute__((ext_vector_type(16))) __bf16 v16bf;
typedef __attribute__((ext_vector_type(8)))  __bf16 v8bf;
typedef __attribute__((ext_vector_type(4)))  __bf16 v4bf;
typedef __attribute__((ext_vector_type(8)))  float  v8f;
typedef __attribute__((ext_vector_type(4)))  float  v4f;
typedef __attribute__((ext_vector_type(4)))  unsigned int u32x4;
typedef __attribute__((ext_vector_type(8)))  int          i32x8;
typedef __attribute__((ext_vector_type(4)))  int          i32x4;

#define LOG2E 1.44269504088896340736f

constexpr int Bb = 8;      // batch
constexpr int L  = 2048;   // sequence
constexpr int D  = 128;    // feature
constexpr int BM = 64;     // query rows per block (16 per wave, 4 waves)
constexpr int BN = 64;     // kv rows per iteration
constexpr int KP = 136;    // Ktile pitch in bf16 (128 + 4-DWORD pad row) -- matches TDM pad
constexpr int VP = 72;     // Vt pitch
constexpr int PP = 72;     // P staging pitch
constexpr int NT = L / BN; // kv iterations

// ---------------------------------------------------------------------------
// WMMA fragment gathers (ISA 7.12.2 layouts)
// A-frag (16x32 bf16 MxK): lane m=l&15 row m; VGPR0-3 = K[h*8, h*8+8),
// VGPR4-7 = K[16+h*8, +8), h=l>>4.  Works on LDS or global bf16.
__device__ __forceinline__ v16bf load_a_frag(const __bf16* base, int pitch, int lane) {
  int m = lane & 15, h = lane >> 4;
  const __bf16* p = base + m * pitch + h * 8;
  union { v16bf v; v8bf q[2]; } u;
  u.q[0] = *(const v8bf*)(p);
  u.q[1] = *(const v8bf*)(p + 16);
  return u.v;
}
// B-frag (32x16 bf16 KxN): lane n=l&15 col n; VGPR0-3 = K[h*8, h*8+8), VGPR4-7 = K[16+h*8, +8).
// Reads row n of the K-major (transposed) operand.
__device__ __forceinline__ v16bf load_b_frag(const __bf16* base, int pitch, int lane) {
  int n = lane & 15, h = lane >> 4;
  const __bf16* p = base + n * pitch + h * 8;
  union { v16bf v; v8bf q[2]; } u;
  u.q[0] = *(const v8bf*)(p);
  u.q[1] = *(const v8bf*)(p + 16);
  return u.v;
}

// ---------------------------------------------------------------------------
// TDM: DMA one BN x D bf16 tile (row-major, row stride D) into LDS with a
// 4-DWORD pad after every row (64 DWORDs) -> LDS pitch 136 bf16.  ISA 8.3/8.4.
__device__ __forceinline__ void tdm_load_tile(const __bf16* gsrc, unsigned lds_off) {
  unsigned long long ga = (unsigned long long)(uintptr_t)gsrc;
  u32x4 g0;
  g0[0] = 1u;                                   // count=1, no gather, user desc
  g0[1] = lds_off;                              // lds_addr (bytes)
  g0[2] = (unsigned)ga;                         // global_addr[31:0]
  g0[3] = (unsigned)((ga >> 32) & 0x1FFFFFFu) | (2u << 30);  // addr[56:32] | type=2
  const unsigned dim0 = 1u << 20, dim1 = 1u << 20;  // generous OOB bounds
  i32x8 g1;
  //        data_size=2B   pad_en      pad_interval=64DW  pad_amount=4DW
  g1[0] = (int)((1u << 16) | (1u << 20) | (5u << 22) | (3u << 25));
  g1[1] = (int)((dim0 & 0xFFFFu) << 16);                          // tensor_dim0 lo16
  g1[2] = (int)(((dim0 >> 16) & 0xFFFFu) | ((dim1 & 0xFFFFu) << 16)); // dim0 hi | dim1 lo
  g1[3] = (int)(((dim1 >> 16) & 0xFFFFu) | ((unsigned)D << 16));  // dim1 hi | tile_dim0=128
  g1[4] = (int)(unsigned)BN;                                      // tile_dim1=64, tile_dim2=0
  g1[5] = (int)(unsigned)D;                                       // tensor_dim0_stride=128
  g1[6] = 0;                                                      // stride hi | dim1_stride
  g1[7] = 0;
  i32x4 gz = {0, 0, 0, 0};
#if __clang_major__ >= 23
  i32x8 gz8 = {0, 0, 0, 0, 0, 0, 0, 0};
  __builtin_amdgcn_tensor_load_to_lds(g0, g1, gz, gz, gz8, 0);
#else
  __builtin_amdgcn_tensor_load_to_lds(g0, g1, gz, gz, 0);
#endif
}

// ---------------------------------------------------------------------------
// One-shot f32 -> bf16 conversion of both inputs into workspace.
__global__ __launch_bounds__(256)
void cvt_bf16_kernel(const float* __restrict__ x1, const float* __restrict__ x2,
                     __bf16* __restrict__ o) {
  const size_t N = (size_t)Bb * L * D;
  size_t i = ((size_t)blockIdx.x * 256 + threadIdx.x) * 4;
  if (i >= 2 * N) return;
  const float* s = (i < N) ? (x1 + i) : (x2 + (i - N));
  v4f f = *(const v4f*)s;
  v4bf b;
#pragma unroll
  for (int e = 0; e < 4; ++e) b[e] = (__bf16)f[e];
  *(v4bf*)(o + i) = b;
}

// ---------------------------------------------------------------------------
// Flash-attention kernel, TDM double-buffered K tiles.
__global__ __launch_bounds__(128)
void fa_tdm_kernel(const float* __restrict__ x1f, const float* __restrict__ x2f,
                   const __bf16* __restrict__ x1b, const __bf16* __restrict__ x2b,
                   float* __restrict__ out) {
  __shared__ __align__(16) __bf16 Ktile[2][BN][KP];   // TDM destination (ping-pong)
  __shared__ __align__(16) __bf16 Vt[D][VP];          // transposed V tile
  __shared__ __align__(16) __bf16 Pst[4][16][PP];     // per-wave P staging

  const int bx   = blockIdx.x;
  const int dir  = bx >> 8;
  const int rem  = bx & 255;
  const int b    = rem >> 5;
  const int tile = rem & 31;

  const __bf16* Qb = (dir ? x2b : x1b) + (size_t)b * L * D;
  const __bf16* Kb = (dir ? x1b : x2b) + (size_t)b * L * D;
  const float*  Qf = (dir ? x2f : x1f) + (size_t)b * L * D;   // verbatim q for epilogue
  float* outg = out + (size_t)dir * Bb * L * 4 * D + (size_t)b * L * 4 * D;

  const int tid  = threadIdx.x;
  const int w    = tid >> 5;
  const int lane = tid & 31;
  const int nidx = lane & 15;
  const int hlf  = lane >> 4;
  const int rowStart = tile * BM + w * 16;

  const unsigned ktile0 = (unsigned)(uintptr_t)&Ktile[0][0][0]; // LDS offset = low 32 bits
  const unsigned ktile1 = (unsigned)(uintptr_t)&Ktile[1][0][0];

  // preload Q fragments straight from bf16 global
  v16bf qf[4];
#pragma unroll
  for (int kk = 0; kk < 4; ++kk)
    qf[kk] = load_a_frag(Qb + (size_t)rowStart * D + kk * 32, D, lane);

  v8f o[8];
#pragma unroll
  for (int df = 0; df < 8; ++df) o[df] = (v8f){0, 0, 0, 0, 0, 0, 0, 0};
  float mrow[8], lrow[8];
#pragma unroll
  for (int r = 0; r < 8; ++r) { mrow[r] = -INFINITY; lrow[r] = 0.0f; }

  if (w == 0) tdm_load_tile(Kb, ktile0);   // prologue: tile 0 in flight

  for (int it = 0; it < NT; ++it) {
    const int buf = it & 1;
    const __bf16* src = Kb + (size_t)it * BN * D;

    __syncthreads();                       // all consumers of both buffers done
    if (w == 0) {
      __builtin_amdgcn_s_wait_tensorcnt(0);         // tile it landed in Ktile[buf]
      if (it + 1 < NT)
        tdm_load_tile(Kb + (size_t)(it + 1) * BN * D, buf ? ktile0 : ktile1);
    }
    __syncthreads();                       // release Ktile[buf] to all waves

    // build transposed V tile from bf16 global (L2-resident; independent of TDM)
    for (int t = tid; t < BN * (D / 8); t += 128) {
      const int row = t >> 4;              // 16 chunks of 8 per row
      const int c8  = (t & 15) * 8;
      v8bf v = *(const v8bf*)(src + row * D + c8);
#pragma unroll
      for (int e = 0; e < 8; ++e) Vt[c8 + e][row] = v[e];
      if (it + 1 < NT)
        __builtin_prefetch(src + BN * D + row * D + c8, 0, 1);
    }

    // ---- S = Q * K^T  (16 x 64 per wave, f32 accum)
    v8f s[4];
#pragma unroll
    for (int nf = 0; nf < 4; ++nf) s[nf] = (v8f){0, 0, 0, 0, 0, 0, 0, 0};
#pragma unroll
    for (int kk = 0; kk < 4; ++kk) {
#pragma unroll
      for (int nf = 0; nf < 4; ++nf) {
        v16bf kb = load_b_frag(&Ktile[buf][nf * 16][kk * 32], KP, lane);
        s[nf] = __builtin_amdgcn_wmma_f32_16x16x32_bf16(
            false, qf[kk], false, kb, (short)0, s[nf], false, false);
      }
    }

    // ---- online softmax (lanes 0-15 own rows 0-7, lanes 16-31 rows 8-15)
    float mnew[8];
#pragma unroll
    for (int r = 0; r < 8; ++r) mnew[r] = mrow[r];
#pragma unroll
    for (int nf = 0; nf < 4; ++nf)
#pragma unroll
      for (int r = 0; r < 8; ++r) mnew[r] = fmaxf(mnew[r], s[nf][r]);
#pragma unroll
    for (int off = 1; off <= 8; off <<= 1)
#pragma unroll
      for (int r = 0; r < 8; ++r)
        mnew[r] = fmaxf(mnew[r], __shfl_xor(mnew[r], off, 32));

    float corr[8];
#pragma unroll
    for (int r = 0; r < 8; ++r) {
      corr[r] = exp2f((mrow[r] - mnew[r]) * LOG2E);
      mrow[r] = mnew[r];
      lrow[r] *= corr[r];
    }
#pragma unroll
    for (int df = 0; df < 8; ++df)
#pragma unroll
      for (int r = 0; r < 8; ++r) o[df][r] *= corr[r];

    float rs[8];
#pragma unroll
    for (int r = 0; r < 8; ++r) rs[r] = 0.0f;
#pragma unroll
    for (int nf = 0; nf < 4; ++nf)
#pragma unroll
      for (int r = 0; r < 8; ++r) {
        float p = exp2f((s[nf][r] - mrow[r]) * LOG2E);
        rs[r] += p;
        Pst[w][hlf * 8 + r][nf * 16 + nidx] = (__bf16)p;   // C-layout -> staging
      }
#pragma unroll
    for (int off = 1; off <= 8; off <<= 1)
#pragma unroll
      for (int r = 0; r < 8; ++r) rs[r] += __shfl_xor(rs[r], off, 32);
#pragma unroll
    for (int r = 0; r < 8; ++r) lrow[r] += rs[r];

    __syncthreads();                       // Vt + Pst visible

    // ---- O += P * V
#pragma unroll
    for (int kt = 0; kt < 2; ++kt) {
      v16bf pa = load_a_frag(&Pst[w][0][kt * 32], PP, lane);
#pragma unroll
      for (int df = 0; df < 8; ++df) {
        v16bf vb = load_b_frag(&Vt[df * 16][kt * 32], VP, lane);
        o[df] = __builtin_amdgcn_wmma_f32_16x16x32_bf16(
            false, pa, false, vb, (short)0, o[df], false, false);
      }
    }
  }

  // ---- epilogue: normalize, fused concat([q, o, q-o, q*o]); q read in f32
  float inv[8];
#pragma unroll
  for (int r = 0; r < 8; ++r) inv[r] = 1.0f / lrow[r];
#pragma unroll
  for (int df = 0; df < 8; ++df) {
#pragma unroll
    for (int r = 0; r < 8; ++r) {
      const int row  = rowStart + hlf * 8 + r;
      const int dcol = df * 16 + nidx;
      float q  = Qf[(size_t)row * D + dcol];
      float ov = o[df][r] * inv[r];
      float* ob = outg + (size_t)row * (4 * D);
      ob[dcol]         = q;
      ob[D + dcol]     = ov;
      ob[2 * D + dcol] = q - ov;
      ob[3 * D + dcol] = q * ov;
    }
  }
}

// ---------------------------------------------------------------------------
// Fallback: self-contained f32-input kernel (no workspace needed).
__global__ __launch_bounds__(128)
void fa_f32_kernel(const float* __restrict__ x1, const float* __restrict__ x2,
                   float* __restrict__ out) {
  __shared__ __align__(16) __bf16 Ktile[BN][KP];
  __shared__ __align__(16) __bf16 Vt[D][VP];
  __shared__ __align__(16) __bf16 Pst[4][16][PP];

  const int bx = blockIdx.x;
  const int dir = bx >> 8, rem = bx & 255, b = rem >> 5, tile = rem & 31;
  const float* Qg = (dir ? x2 : x1) + (size_t)b * L * D;
  const float* Kg = (dir ? x1 : x2) + (size_t)b * L * D;
  float* outg = out + (size_t)dir * Bb * L * 4 * D + (size_t)b * L * 4 * D;

  const int tid = threadIdx.x, w = tid >> 5, lane = tid & 31;
  const int nidx = lane & 15, hlf = lane >> 4;
  const int rowStart = tile * BM + w * 16;

  v16bf qf[4];
#pragma unroll
  for (int kk = 0; kk < 4; ++kk) {
    const float* qp = Qg + (size_t)(rowStart + nidx) * D + kk * 32 + hlf * 8;
    v4f f0 = *(const v4f*)(qp);
    v4f f1 = *(const v4f*)(qp + 4);
    v4f f2 = *(const v4f*)(qp + 16);
    v4f f3 = *(const v4f*)(qp + 20);
    v16bf q;
#pragma unroll
    for (int e = 0; e < 4; ++e) {
      q[e] = (__bf16)f0[e]; q[4 + e] = (__bf16)f1[e];
      q[8 + e] = (__bf16)f2[e]; q[12 + e] = (__bf16)f3[e];
    }
    qf[kk] = q;
  }

  v8f o[8];
#pragma unroll
  for (int df = 0; df < 8; ++df) o[df] = (v8f){0, 0, 0, 0, 0, 0, 0, 0};
  float mrow[8], lrow[8];
#pragma unroll
  for (int r = 0; r < 8; ++r) { mrow[r] = -INFINITY; lrow[r] = 0.0f; }

  for (int it = 0; it < NT; ++it) {
    const float* src = Kg + (size_t)it * BN * D;
    __syncthreads();
    for (int t = tid; t < BN * (D / 4); t += 128) {
      const int row = t >> 5, c4 = (t & 31) * 4;
      v4f f = *(const v4f*)(src + row * D + c4);
#pragma unroll
      for (int e = 0; e < 4; ++e) {
        __bf16 bv = (__bf16)f[e];
        Ktile[row][c4 + e] = bv;
        Vt[c4 + e][row] = bv;
      }
      if (it + 1 < NT) __builtin_prefetch(src + BN * D + row * D + c4, 0, 1);
    }
    __syncthreads();

    v8f s[4];
#pragma unroll
    for (int nf = 0; nf < 4; ++nf) s[nf] = (v8f){0, 0, 0, 0, 0, 0, 0, 0};
#pragma unroll
    for (int kk = 0; kk < 4; ++kk)
#pragma unroll
      for (int nf = 0; nf < 4; ++nf) {
        v16bf kb = load_b_frag(&Ktile[nf * 16][kk * 32], KP, lane);
        s[nf] = __builtin_amdgcn_wmma_f32_16x16x32_bf16(
            false, qf[kk], false, kb, (short)0, s[nf], false, false);
      }

    float mnew[8];
#pragma unroll
    for (int r = 0; r < 8; ++r) mnew[r] = mrow[r];
#pragma unroll
    for (int nf = 0; nf < 4; ++nf)
#pragma unroll
      for (int r = 0; r < 8; ++r) mnew[r] = fmaxf(mnew[r], s[nf][r]);
#pragma unroll
    for (int off = 1; off <= 8; off <<= 1)
#pragma unroll
      for (int r = 0; r < 8; ++r)
        mnew[r] = fmaxf(mnew[r], __shfl_xor(mnew[r], off, 32));
    float corr[8];
#pragma unroll
    for (int r = 0; r < 8; ++r) {
      corr[r] = exp2f((mrow[r] - mnew[r]) * LOG2E);
      mrow[r] = mnew[r];
      lrow[r] *= corr[r];
    }
#pragma unroll
    for (int df = 0; df < 8; ++df)
#pragma unroll
      for (int r = 0; r < 8; ++r) o[df][r] *= corr[r];
    float rs[8];
#pragma unroll
    for (int r = 0; r < 8; ++r) rs[r] = 0.0f;
#pragma unroll
    for (int nf = 0; nf < 4; ++nf)
#pragma unroll
      for (int r = 0; r < 8; ++r) {
        float p = exp2f((s[nf][r] - mrow[r]) * LOG2E);
        rs[r] += p;
        Pst[w][hlf * 8 + r][nf * 16 + nidx] = (__bf16)p;
      }
#pragma unroll
    for (int off = 1; off <= 8; off <<= 1)
#pragma unroll
      for (int r = 0; r < 8; ++r) rs[r] += __shfl_xor(rs[r], off, 32);
#pragma unroll
    for (int r = 0; r < 8; ++r) lrow[r] += rs[r];

    __syncthreads();
#pragma unroll
    for (int kt = 0; kt < 2; ++kt) {
      v16bf pa = load_a_frag(&Pst[w][0][kt * 32], PP, lane);
#pragma unroll
      for (int df = 0; df < 8; ++df) {
        v16bf vb = load_b_frag(&Vt[df * 16][kt * 32], VP, lane);
        o[df] = __builtin_amdgcn_wmma_f32_16x16x32_bf16(
            false, pa, false, vb, (short)0, o[df], false, false);
      }
    }
  }

  float inv[8];
#pragma unroll
  for (int r = 0; r < 8; ++r) inv[r] = 1.0f / lrow[r];
#pragma unroll
  for (int df = 0; df < 8; ++df)
#pragma unroll
    for (int r = 0; r < 8; ++r) {
      const int row = rowStart + hlf * 8 + r;
      const int dcol = df * 16 + nidx;
      float q = Qg[(size_t)row * D + dcol];
      float ov = o[df][r] * inv[r];
      float* ob = outg + (size_t)row * (4 * D);
      ob[dcol] = q;
      ob[D + dcol] = ov;
      ob[2 * D + dcol] = q - ov;
      ob[3 * D + dcol] = q * ov;
    }
}

extern "C" void kernel_launch(void* const* d_in, const int* in_sizes, int n_in,
                              void* d_out, int out_size, void* d_ws, size_t ws_size,
                              hipStream_t stream) {
  (void)in_sizes; (void)n_in; (void)out_size;
  const float* x1 = (const float*)d_in[0];
  const float* x2 = (const float*)d_in[1];
  float* out = (float*)d_out;

  const size_t N = (size_t)Bb * L * D;            // elements per input
  const size_t need = 2 * N * 2;                  // bf16 copies of x1 and x2
  dim3 grid(2 * Bb * (L / BM));                   // 512 blocks, 4 wave32s each

  if (d_ws != nullptr && ws_size >= need) {
    __bf16* wb = (__bf16*)d_ws;
    int cvt_blocks = (int)((2 * N / 4 + 255) / 256);
    cvt_bf16_kernel<<<cvt_blocks, 256, 0, stream>>>(x1, x2, wb);
    fa_tdm_kernel<<<grid, 128, 0, stream>>>(x1, x2, wb, wb + N, out);
  } else {
    fa_f32_kernel<<<grid, 128, 0, stream>>>(x1, x2, out);
  }
}
